// SparseMoE_76218489635336
// MI455X (gfx1250) — matrix operand order repaired and checked by reading, not compiled, not run on hardware
//
#include <hip/hip_runtime.h>
#include <hip/hip_bf16.h>
#include <math.h>

// Problem constants (from reference)
#define F_TOT 8192      // B*L tokens
#define D_DIM 1024
#define E_NUM 8
#define FF_DIM 4096
#define CAP 2560        // int(F_TOT * K/E * 1.25)

typedef __bf16 bf16;
typedef __attribute__((ext_vector_type(16))) __bf16 v16bf;
typedef __attribute__((ext_vector_type(8)))  __bf16 v8bf;
typedef __attribute__((ext_vector_type(2)))  __bf16 v2bf;
typedef __attribute__((ext_vector_type(8)))  float  v8f;

union Frag16 { v16bf v; v8bf h[2]; };
union Pack16 { bf16 a[16]; v8bf v[2]; };

__device__ __forceinline__ v16bf load_frag(const bf16* p_lo, const bf16* p_hi) {
    Frag16 f;
    f.h[0] = *(const v8bf*)p_lo;   // ds_load_b128
    f.h[1] = *(const v8bf*)p_hi;   // ds_load_b128
    return f.v;
}

// ---------------------------------------------------------------------------
// Router: logits = x@Wg+bg ; noisy = logits + noise*softplus(x@Wn+bn);
// top-2 + softmax. One wave per token.
// ---------------------------------------------------------------------------
__global__ __launch_bounds__(256)
void moe_router(const float* __restrict__ X, const float* __restrict__ noise,
                const float* __restrict__ Wg, const float* __restrict__ bg,
                const float* __restrict__ Wn, const float* __restrict__ bn,
                int* __restrict__ topidx, float* __restrict__ topgate)
{
    const int lane = threadIdx.x & 31;
    const int wid  = threadIdx.x >> 5;
    const int t    = blockIdx.x * 8 + wid;     // token index (grid = F/8)

    const float* xr = X + (size_t)t * D_DIM;
    float ag[E_NUM], an[E_NUM];
    #pragma unroll
    for (int e = 0; e < E_NUM; ++e) { ag[e] = 0.f; an[e] = 0.f; }

    for (int d = lane; d < D_DIM; d += 32) {
        const float xv = xr[d];
        const float* wg = Wg + (size_t)d * E_NUM;
        const float* wn = Wn + (size_t)d * E_NUM;
        #pragma unroll
        for (int e = 0; e < E_NUM; ++e) {
            ag[e] += xv * wg[e];
            an[e] += xv * wn[e];
        }
    }
    #pragma unroll
    for (int e = 0; e < E_NUM; ++e) {
        #pragma unroll
        for (int off = 16; off > 0; off >>= 1) {
            ag[e] += __shfl_xor(ag[e], off, 32);
            an[e] += __shfl_xor(an[e], off, 32);
        }
    }
    if (lane == 0) {
        float noisy[E_NUM];
        #pragma unroll
        for (int e = 0; e < E_NUM; ++e) {
            const float lg = ag[e] + bg[e];
            const float nl = an[e] + bn[e];
            const float sp = (nl > 20.f) ? nl : log1pf(expf(nl));   // softplus
            noisy[e] = lg + noise[(size_t)t * E_NUM + e] * sp;
        }
        int i0 = 0; float v0 = noisy[0];
        #pragma unroll
        for (int e = 1; e < E_NUM; ++e) if (noisy[e] > v0) { v0 = noisy[e]; i0 = e; }
        int i1 = -1; float v1 = -3.4e38f;
        #pragma unroll
        for (int e = 0; e < E_NUM; ++e) if (e != i0 && noisy[e] > v1) { v1 = noisy[e]; i1 = e; }
        const float m  = fmaxf(v0, v1);
        const float e0 = expf(v0 - m), e1 = expf(v1 - m);
        const float inv = 1.0f / (e0 + e1);
        topidx [t * 2 + 0] = i0;  topidx [t * 2 + 1] = i1;
        topgate[t * 2 + 0] = e0 * inv; topgate[t * 2 + 1] = e1 * inv;
    }
}

// ---------------------------------------------------------------------------
// Dispatch build: one wave per expert; token-order prefix via ballot+popc
// (deterministic; mirrors the reference cumsum+capacity drop).
// Records the inverse map slotmap[t][k] for the gather-based combine.
// ---------------------------------------------------------------------------
__global__ __launch_bounds__(256)
void moe_dispatch(const int* __restrict__ topidx,
                  int* __restrict__ disp, int* __restrict__ slotmap)
{
    const int lane = threadIdx.x & 31;
    const int e    = threadIdx.x >> 5;
    int* de = disp + e * CAP;
    for (int i = lane; i < CAP; i += 32) de[i] = F_TOT;   // invalid marker

    int base = 0;
    for (int t0 = 0; t0 < F_TOT; t0 += 32) {
        const int t  = t0 + lane;
        const int i0 = topidx[t * 2 + 0];
        const int i1 = topidx[t * 2 + 1];
        const bool hit0 = (i0 == e), hit1 = (i1 == e);
        const bool pred = hit0 || hit1;
        const unsigned long long mask = __ballot(pred);
        const int prefix = __popcll(mask & ((1ull << lane) - 1ull));
        if (pred) {
            const int pos = base + prefix;
            const int kk  = hit0 ? 0 : 1;
            if (pos < CAP) { de[pos] = t; slotmap[t * 2 + kk] = pos; }
            else           { slotmap[t * 2 + kk] = -1; }          // dropped
        }
        base += (int)__popcll(mask);
    }
}

// ---------------------------------------------------------------------------
// Tiled bf16 WMMA GEMM, 128x128 tile, BK=32, 8 waves, each wave = 64x32.
// Fragment layouts per CDNA5 ISA 7.12.2 (wave32).
// Software-pipelined: next tile's globals prefetched into registers while
// the current tile's WMMAs issue.
// ---------------------------------------------------------------------------
#define BM 128
#define BN 128
#define BK 32
#define LDA 40   // padded LDS row stride (bf16 elems); 80B keeps b128 alignment

// GEMM1: H[e] = gelu(gather(X, dispatch[e]) @ W1[e] + b1[e])   (bf16 out)
__global__ __launch_bounds__(256)
void moe_ffn1(const float* __restrict__ X,       // [F,D]
              const float* __restrict__ W1,      // [E,D,FF]
              const float* __restrict__ b1,      // [E,FF]
              const int*   __restrict__ disp,    // [E,CAP]
              bf16*        __restrict__ H)       // [E,CAP,FF]
{
    __shared__ bf16 As[BM * LDA];
    __shared__ bf16 Bs[BN * LDA];   // transposed: Bs[n][k]

    const int e  = blockIdx.z;
    const int m0 = blockIdx.y * BM;
    const int n0 = blockIdx.x * BN;

    const int tid  = threadIdx.x;
    const int lane = tid & 31;
    const int wid  = tid >> 5;
    const int wm = (wid >> 2) * 64;
    const int wn = (wid & 3) * 32;

    const float* W1e = W1 + (size_t)e * D_DIM * FF_DIM;
    const int*   de  = disp + e * CAP;
    bf16*        He  = H + (size_t)e * CAP * FF_DIM;

    // A staging coords: 128 rows x 2 half-rows of 16 floats
    const int arow = tid >> 1;
    const int aseg = (tid & 1) * 16;
    const int srow = de[m0 + arow];
    const bool valid = (srow < F_TOT);
    const float* xrow = X + (size_t)(valid ? srow : 0) * D_DIM;

    // B staging coords: each thread owns a k-pair for 8 n values
    const int pk  = (tid >> 4) * 2;      // k-pair base: 0,2,...,30
    const int nb8 = (tid & 15) * 8;      // n base: 0..120

    const v8f zacc = {0.f,0.f,0.f,0.f,0.f,0.f,0.f,0.f};
    v8f acc[4][2];
    #pragma unroll
    for (int i = 0; i < 4; ++i)
        #pragma unroll
        for (int j = 0; j < 2; ++j) acc[i][j] = zacc;

    const int lm  = lane & 15;
    const int alo = (lane >> 4) * 8;    // A: K base per ISA layout (0 / 8)
    const int blo = (lane >> 4) * 8;    // B: K base per ISA layout (0 / 8)

    float ra[16];          // prefetched A floats
    float rb0[8], rb1[8];  // prefetched B floats (rows pk, pk+1)

    auto prefetch = [&](int k0) {
        if (valid) {
            const float4* p = (const float4*)(xrow + k0 + aseg);
            #pragma unroll
            for (int q = 0; q < 4; ++q) {
                const float4 f = p[q];
                ra[q*4+0] = f.x; ra[q*4+1] = f.y; ra[q*4+2] = f.z; ra[q*4+3] = f.w;
            }
        } else {
            #pragma unroll
            for (int q = 0; q < 16; ++q) ra[q] = 0.f;
        }
        const float4* q0 = (const float4*)(W1e + (size_t)(k0 + pk)     * FF_DIM + n0 + nb8);
        const float4* q1 = (const float4*)(W1e + (size_t)(k0 + pk + 1) * FF_DIM + n0 + nb8);
        #pragma unroll
        for (int q = 0; q < 2; ++q) {
            const float4 f0 = q0[q], f1 = q1[q];
            rb0[q*4+0]=f0.x; rb0[q*4+1]=f0.y; rb0[q*4+2]=f0.z; rb0[q*4+3]=f0.w;
            rb1[q*4+0]=f1.x; rb1[q*4+1]=f1.y; rb1[q*4+2]=f1.z; rb1[q*4+3]=f1.w;
        }
    };

    prefetch(0);
    for (int k0 = 0; k0 < D_DIM; k0 += BK) {
        __syncthreads();
        {   // stage A: 16 bf16 contiguous (2 x ds_store_b128)
            Pack16 u;
            #pragma unroll
            for (int q = 0; q < 16; ++q) u.a[q] = (bf16)ra[q];
            bf16* dst = As + arow * LDA + aseg;
            *(v8bf*)(dst)     = u.v[0];
            *(v8bf*)(dst + 8) = u.v[1];
        }
        {   // stage B transposed, packed k-pairs (8 x ds_store_b32)
            #pragma unroll
            for (int j = 0; j < 8; ++j) {
                v2bf p; p[0] = (bf16)rb0[j]; p[1] = (bf16)rb1[j];
                *(v2bf*)(&Bs[(nb8 + j) * LDA + pk]) = p;
            }
        }
        __syncthreads();

        if (k0 + BK < D_DIM) prefetch(k0 + BK);   // overlaps with WMMAs below

        v16bf af[4], bfr[2];
        #pragma unroll
        for (int i = 0; i < 4; ++i) {
            const bf16* base = As + (wm + i*16 + lm) * LDA;
            af[i] = load_frag(base + alo, base + alo + 16);
        }
        #pragma unroll
        for (int j = 0; j < 2; ++j) {
            const bf16* base = Bs + (wn + j*16 + lm) * LDA;
            bfr[j] = load_frag(base + blo, base + blo + 16);
        }
        #pragma unroll
        for (int i = 0; i < 4; ++i)
            #pragma unroll
            for (int j = 0; j < 2; ++j)
                acc[i][j] = __builtin_amdgcn_wmma_f32_16x16x32_bf16(
                    false, af[i], false, bfr[j], (short)0, acc[i][j], false, false);
    }

    // Epilogue: bias + exact GELU -> bf16 H
    const float* b1e = b1 + e * FF_DIM;
    #pragma unroll
    for (int j = 0; j < 2; ++j) {
        const int n = n0 + wn + j*16 + lm;
        const float bias = b1e[n];
        #pragma unroll
        for (int i = 0; i < 4; ++i) {
            #pragma unroll
            for (int r = 0; r < 8; ++r) {
                const int m = m0 + wm + i*16 + (lane >> 4) * 8 + r;
                float v = acc[i][j][r] + bias;
                v = 0.5f * v * (1.0f + erff(v * 0.70710678118654752f));
                He[(size_t)m * FF_DIM + n] = (bf16)v;
            }
        }
    }
}

// GEMM2: Y[e] = H[e] @ W2[e] + b2[e]   (fp32 out, gate applied in combine)
__global__ __launch_bounds__(256)
void moe_ffn2(const bf16*  __restrict__ H,       // [E,CAP,FF] bf16
              const float* __restrict__ W2,      // [E,FF,D]
              const float* __restrict__ b2,      // [E,D]
              float*       __restrict__ Y)       // [E,CAP,D]
{
    __shared__ bf16 As[BM * LDA];
    __shared__ bf16 Bs[BN * LDA];

    const int e  = blockIdx.z;
    const int m0 = blockIdx.y * BM;
    const int n0 = blockIdx.x * BN;

    const int tid  = threadIdx.x;
    const int lane = tid & 31;
    const int wid  = tid >> 5;
    const int wm = (wid >> 2) * 64;
    const int wn = (wid & 3) * 32;

    const bf16*  He  = H + (size_t)e * CAP * FF_DIM;
    const float* W2e = W2 + (size_t)e * FF_DIM * D_DIM;
    float*       Ye  = Y + (size_t)e * CAP * D_DIM;

    const int arow = tid >> 1;
    const int aseg = (tid & 1) * 16;
    const int pk  = (tid >> 4) * 2;
    const int nb8 = (tid & 15) * 8;

    const v8f zacc = {0.f,0.f,0.f,0.f,0.f,0.f,0.f,0.f};
    v8f acc[4][2];
    #pragma unroll
    for (int i = 0; i < 4; ++i)
        #pragma unroll
        for (int j = 0; j < 2; ++j) acc[i][j] = zacc;

    const int lm  = lane & 15;
    const int alo = (lane >> 4) * 8;
    const int blo = (lane >> 4) * 8;

    v8bf ra0, ra1;         // prefetched A (already bf16)
    float rb0[8], rb1[8];  // prefetched B floats

    auto prefetch = [&](int k0) {
        const bf16* hp = He + (size_t)(m0 + arow) * FF_DIM + k0 + aseg;
        ra0 = *(const v8bf*)(hp);
        ra1 = *(const v8bf*)(hp + 8);
        const float4* q0 = (const float4*)(W2e + (size_t)(k0 + pk)     * D_DIM + n0 + nb8);
        const float4* q1 = (const float4*)(W2e + (size_t)(k0 + pk + 1) * D_DIM + n0 + nb8);
        #pragma unroll
        for (int q = 0; q < 2; ++q) {
            const float4 f0 = q0[q], f1 = q1[q];
            rb0[q*4+0]=f0.x; rb0[q*4+1]=f0.y; rb0[q*4+2]=f0.z; rb0[q*4+3]=f0.w;
            rb1[q*4+0]=f1.x; rb1[q*4+1]=f1.y; rb1[q*4+2]=f1.z; rb1[q*4+3]=f1.w;
        }
    };

    prefetch(0);
    for (int k0 = 0; k0 < FF_DIM; k0 += BK) {
        __syncthreads();
        {   // stage A (bf16 direct, 2 x ds_store_b128)
            bf16* dst = As + arow * LDA + aseg;
            *(v8bf*)(dst)     = ra0;
            *(v8bf*)(dst + 8) = ra1;
        }
        {   // stage B transposed, packed k-pairs
            #pragma unroll
            for (int j = 0; j < 8; ++j) {
                v2bf p; p[0] = (bf16)rb0[j]; p[1] = (bf16)rb1[j];
                *(v2bf*)(&Bs[(nb8 + j) * LDA + pk]) = p;
            }
        }
        __syncthreads();

        if (k0 + BK < FF_DIM) prefetch(k0 + BK);

        v16bf af[4], bfr[2];
        #pragma unroll
        for (int i = 0; i < 4; ++i) {
            const bf16* base = As + (wm + i*16 + lm) * LDA;
            af[i] = load_frag(base + alo, base + alo + 16);
        }
        #pragma unroll
        for (int j = 0; j < 2; ++j) {
            const bf16* base = Bs + (wn + j*16 + lm) * LDA;
            bfr[j] = load_frag(base + blo, base + blo + 16);
        }
        #pragma unroll
        for (int i = 0; i < 4; ++i)
            #pragma unroll
            for (int j = 0; j < 2; ++j)
                acc[i][j] = __builtin_amdgcn_wmma_f32_16x16x32_bf16(
                    false, af[i], false, bfr[j], (short)0, acc[i][j], false, false);
    }

    const float* b2e = b2 + e * D_DIM;
    float bias[2];
    #pragma unroll
    for (int j = 0; j < 2; ++j) bias[j] = b2e[n0 + wn + j*16 + lm];
    #pragma unroll
    for (int i = 0; i < 4; ++i) {
        #pragma unroll
        for (int r = 0; r < 8; ++r) {
            const int m = m0 + wm + i*16 + (lane >> 4) * 8 + r;
            #pragma unroll
            for (int j = 0; j < 2; ++j) {
                const int n = n0 + wn + j*16 + lm;
                Ye[(size_t)m * D_DIM + n] = acc[i][j][r] + bias[j];
            }
        }
    }
}

// ---------------------------------------------------------------------------
// Combine: out[t] = sum_k gate[t][k] * Y[expert[t][k]][slot[t][k]]
// Gather-based -> no atomics, every output element written exactly once.
// ---------------------------------------------------------------------------
__global__ __launch_bounds__(256)
void moe_combine(const float* __restrict__ Y,        // [E,CAP,D]
                 const int*   __restrict__ topidx,   // [F,2]
                 const float* __restrict__ topgate,  // [F,2]
                 const int*   __restrict__ slotmap,  // [F,2]
                 float*       __restrict__ out)      // [F,D]
{
    const int t  = blockIdx.x;
    const int d4 = threadIdx.x;
    const int e0 = topidx[t*2+0], e1 = topidx[t*2+1];
    const int s0 = slotmap[t*2+0], s1 = slotmap[t*2+1];
    const float g0 = topgate[t*2+0], g1 = topgate[t*2+1];

    float4 r = {0.f, 0.f, 0.f, 0.f};
    if (s0 >= 0) {
        const float4 y = ((const float4*)(Y + ((size_t)e0 * CAP + s0) * D_DIM))[d4];
        r.x += g0*y.x; r.y += g0*y.y; r.z += g0*y.z; r.w += g0*y.w;
    }
    if (s1 >= 0) {
        const float4 y = ((const float4*)(Y + ((size_t)e1 * CAP + s1) * D_DIM))[d4];
        r.x += g1*y.x; r.y += g1*y.y; r.z += g1*y.z; r.w += g1*y.w;
    }
    ((float4*)(out + (size_t)t * D_DIM))[d4] = r;
}

// ---------------------------------------------------------------------------
extern "C" void kernel_launch(void* const* d_in, const int* in_sizes, int n_in,
                              void* d_out, int out_size, void* d_ws, size_t ws_size,
                              hipStream_t stream) {
    const float* x     = (const float*)d_in[0];
    const float* noise = (const float*)d_in[1];
    const float* Wg    = (const float*)d_in[2];
    const float* bg    = (const float*)d_in[3];
    const float* Wn    = (const float*)d_in[4];
    const float* bn    = (const float*)d_in[5];
    const float* W1    = (const float*)d_in[6];
    const float* b1    = (const float*)d_in[7];
    const float* W2    = (const float*)d_in[8];
    const float* b2    = (const float*)d_in[9];
    float* out = (float*)d_out;

    // Workspace layout (total ~240.4 MB):
    //   topidx  : F*2 int        =  64 KB
    //   topgate : F*2 float      =  64 KB
    //   slotmap : F*2 int        =  64 KB
    //   dispatch: E*CAP int      =  80 KB
    //   H       : E*CAP*FF bf16  = 160 MB
    //   Y       : E*CAP*D  fp32  =  80 MB
    char* ws = (char*)d_ws;
    size_t off = 0;
    auto alloc = [&](size_t bytes) -> void* {
        void* p = ws + off;
        off += (bytes + 255) & ~(size_t)255;
        return p;
    };
    int*   topidx  = (int*)  alloc((size_t)F_TOT * 2 * sizeof(int));
    float* topgate = (float*)alloc((size_t)F_TOT * 2 * sizeof(float));
    int*   slotmap = (int*)  alloc((size_t)F_TOT * 2 * sizeof(int));
    int*   disp    = (int*)  alloc((size_t)E_NUM * CAP * sizeof(int));
    bf16*  H       = (bf16*) alloc((size_t)E_NUM * CAP * FF_DIM * sizeof(bf16));
    float* Y       = (float*)alloc((size_t)E_NUM * CAP * D_DIM * sizeof(float));

    moe_router  <<<F_TOT / 8, 256, 0, stream>>>(x, noise, Wg, bg, Wn, bn, topidx, topgate);
    moe_dispatch<<<1, 256, 0, stream>>>(topidx, disp, slotmap);
    moe_ffn1    <<<dim3(FF_DIM / BN, CAP / BM, E_NUM), 256, 0, stream>>>(x, W1, b1, disp, H);
    moe_ffn2    <<<dim3(D_DIM / BN, CAP / BM, E_NUM), 256, 0, stream>>>(H, W2, b2, Y);
    moe_combine <<<F_TOT, 256, 0, stream>>>(Y, topidx, topgate, slotmap, out);
}
